// DenoisingModule_3822520893491
// MI455X (gfx1250) — matrix operand order repaired and checked by reading, not compiled, not run on hardware
//
#include <hip/hip_runtime.h>

typedef __attribute__((ext_vector_type(16))) __bf16 v16bf;
typedef __attribute__((ext_vector_type(8)))  float  v8f;

__device__ __forceinline__ unsigned short f2bf(float f) {
  union { float f; unsigned u; } v; v.f = f;
  unsigned r = v.u + 0x7FFFu + ((v.u >> 16) & 1u);   // round-to-nearest-even
  return (unsigned short)(r >> 16);
}

__device__ __forceinline__ v8f wmma_bf16(v16bf a, v16bf b, v8f c) {
  return __builtin_amdgcn_wmma_f32_16x16x32_bf16(false, a, false, b, (short)0, c, false, false);
}

// ---------------------------------------------------------------------------
// Fragment layouts (CDNA5 ISA 7.12.2, 16-bit 16x32 A / 32x16 B tiles):
// A element (M, K):  lane = M + 16*((K>>3)&1); vgpr = 4*(K>>4) + ((K&7)>>1); half = K&1
// B element (K, N):  lane = N + 16*((K>>3)&1); vgpr = 4*(K>>4) + ((K&7)>>1); half = K&1
// Tiles stored as 512 bf16: flat = lane*16 + vgpr*2 + half -> per-lane v16bf load.
//
// K2 computes f^T = phi(A) x theta(B): its C-fragment (M=m in regs, N=n in lanes)
// is layout-identical to the den-GEMM A-fragment, so fnorm needs NO transpose.
// ---------------------------------------------------------------------------

// ============================================================================
// K1a: theta/phi projections -> fragment-packed bf16
//   phi   -> A frags (M=m, K=d): phF[((b*256+mtile)*4+kt)*512 + flatA]
//   theta -> B frags (K=d, N=n): thF[((b*256+ntile)*4+kt)*512 + flatB]
// ============================================================================
__global__ __launch_bounds__(256) void k1_proj(
    const float* __restrict__ x,
    const float* __restrict__ w_theta, const float* __restrict__ b_theta,
    const float* __restrict__ w_phi,   const float* __restrict__ b_phi,
    unsigned short* __restrict__ thF,  unsigned short* __restrict__ phF)
{
  __shared__ float xs[32 * 260];   // [n][c], row stride 260 floats (bank padding)
  const int t = threadIdx.x;
  const int b  = blockIdx.x >> 7;
  const int n0 = (blockIdx.x & 127) * 32;

  #pragma unroll
  for (int i = 0; i < 8; ++i) {
    int c = (t >> 3) + i * 32;
    int n = (t & 7) * 4;
    float4 v = *(const float4*)(x + ((size_t)(b * 256 + c) * 4096 + n0 + n));
    xs[(n + 0) * 260 + c] = v.x;
    xs[(n + 1) * 260 + c] = v.y;
    xs[(n + 2) * 260 + c] = v.z;
    xs[(n + 3) * 260 + c] = v.w;
  }
  __syncthreads();

  const int wv = t >> 5, lane = t & 31;
  const float4* xv = (const float4*)(xs + lane * 260);

  for (int i = 0; i < 16; ++i) {
    const int d = wv + i * 8;
    const float4* wt = (const float4*)(w_theta + d * 256);
    const float4* wp = (const float4*)(w_phi   + d * 256);
    float at = b_theta[d], ap = b_phi[d];
    #pragma unroll 8
    for (int j = 0; j < 64; ++j) {
      float4 xx = xv[j];
      float4 a  = wt[j];
      float4 p  = wp[j];
      at += a.x * xx.x + a.y * xx.y + a.z * xx.z + a.w * xx.w;
      ap += p.x * xx.x + p.y * xx.y + p.z * xx.z + p.w * xx.w;
    }
    const int n = n0 + lane;
    const int ntile = n >> 4, M = n & 15;
    const int kt = d >> 5, kk = d & 31;
    // theta -> B frag (K=d, N=n)
    {
      int laneF = M + 16 * ((kk >> 3) & 1);
      int vg    = 4 * (kk >> 4) + ((kk & 7) >> 1);
      thF[(size_t)(((b * 256 + ntile) * 4 + kt)) * 512 + laneF * 16 + vg * 2 + (kk & 1)] = f2bf(at);
    }
    // phi -> A frag (M=m, K=d)
    {
      int laneF = M + 16 * ((kk >> 3) & 1);
      int vg    = 4 * (kk >> 4) + ((kk & 7) >> 1);
      phF[(size_t)(((b * 256 + ntile) * 4 + kt)) * 512 + laneF * 16 + vg * 2 + (kk & 1)] = f2bf(ap);
    }
  }
}

// ============================================================================
// K1b: x^T -> B-fragment bf16 (K=m tiles of 32, N=c tiles of 16)
// ============================================================================
__global__ __launch_bounds__(256) void k1_xt(const float* __restrict__ x,
                                             unsigned short* __restrict__ xtF)
{
  const int o  = blockIdx.x * 256 + threadIdx.x;
  const int mg = o & 255;
  const int c  = (o >> 8) & 255;
  const int b  = o >> 16;
  const float4* xp = (const float4*)(x + ((size_t)(b * 256 + c) * 4096 + mg * 16));
  float4 v0 = xp[0], v1 = xp[1], v2 = xp[2], v3 = xp[3];
  const int ct = c >> 4, mk = mg >> 1, hi = mg & 1;
  unsigned* dst = (unsigned*)(xtF + (size_t)(((b * 16 + ct) * 128 + mk)) * 512
                              + (c & 15) * 16 + 8 * hi);
  dst[0] = (unsigned)f2bf(v0.x) | ((unsigned)f2bf(v0.y) << 16);
  dst[1] = (unsigned)f2bf(v0.z) | ((unsigned)f2bf(v0.w) << 16);
  dst[2] = (unsigned)f2bf(v1.x) | ((unsigned)f2bf(v1.y) << 16);
  dst[3] = (unsigned)f2bf(v1.z) | ((unsigned)f2bf(v1.w) << 16);
  dst[128] = (unsigned)f2bf(v2.x) | ((unsigned)f2bf(v2.y) << 16);
  dst[129] = (unsigned)f2bf(v2.z) | ((unsigned)f2bf(v2.w) << 16);
  dst[130] = (unsigned)f2bf(v3.x) | ((unsigned)f2bf(v3.y) << 16);
  dst[131] = (unsigned)f2bf(v3.z) | ((unsigned)f2bf(v3.w) << 16);
}

// ============================================================================
// K1c: w_conv -> A-fragment bf16
// ============================================================================
__global__ __launch_bounds__(256) void k1_wc(const float* __restrict__ w_conv,
                                             unsigned short* __restrict__ wcF)
{
  const int o  = blockIdx.x * 256 + threadIdx.x;
  const int c0 = (o & 31) * 8;
  const int d  = o >> 5;
  const float4* wp = (const float4*)(w_conv + (size_t)d * 256 + c0);
  float4 v0 = wp[0], v1 = wp[1];
  const int kk = c0 & 31;
  const int laneF = (d & 15) + 16 * ((kk >> 3) & 1);
  const int g = kk >> 4;
  unsigned* dst = (unsigned*)(wcF + (size_t)(((d >> 4) * 8 + (c0 >> 5))) * 512
                              + laneF * 16 + 8 * g);
  dst[0] = (unsigned)f2bf(v0.x) | ((unsigned)f2bf(v0.y) << 16);
  dst[1] = (unsigned)f2bf(v0.z) | ((unsigned)f2bf(v0.w) << 16);
  dst[2] = (unsigned)f2bf(v1.x) | ((unsigned)f2bf(v1.y) << 16);
  dst[3] = (unsigned)f2bf(v1.z) | ((unsigned)f2bf(v1.w) << 16);
}

// ============================================================================
// K2: fused  f^T = phi^T theta / 16 ; softmax over batch ; den = f x^T
//   8 waves = 8 batches; one 16-row n-tile per block; m-tiles of 32.
//   LDS exchange carries exp values only; reciprocal sums computed once.
// ============================================================================
__global__ __launch_bounds__(256, 1) void k2_attn(
    const unsigned short* __restrict__ phF,   // phi A-frags
    const unsigned short* __restrict__ thF,   // theta B-frags
    const unsigned short* __restrict__ xtF,   // x^T B-frags
    float* __restrict__ den_nc)
{
  __shared__ float ebuf[512 * 10 + 8];  // [elem(m*16+n)]*10 + b
  __shared__ float sbuf[16 * 34];       // [n]*34 + m : 1/sum_b

  const int t = threadIdx.x, b = t >> 5, lane = t & 31;
  const int hiL = lane >> 4, lm = lane & 15;
  const int ntile = blockIdx.x;

  // theta B fragments: resident all kernel
  v16bf thB[4];
  #pragma unroll
  for (int kt = 0; kt < 4; ++kt)
    thB[kt] = *(const v16bf*)(thF + (size_t)(((b * 256 + ntile) * 4 + kt)) * 512 + lane * 16);

  v8f dacc[16];
  {
    v8f zz = {};
    #pragma unroll
    for (int ct = 0; ct < 16; ++ct) dacc[ct] = zz;
  }

  const unsigned short* phB = phF + (size_t)b * 256 * 4 * 512 + lane * 16;
  const unsigned short* xtB = xtF + (size_t)b * 16 * 128 * 512 + lane * 16;

  // preload phi A-frags for (mi=0, s=0)
  v16bf pa[4];
  #pragma unroll
  for (int kt = 0; kt < 4; ++kt)
    pa[kt] = *(const v16bf*)(phB + (size_t)kt * 512);

  for (int mi = 0; mi < 128; ++mi) {
    // ---- issue s=1 loads, then f^T WMMAs (partial waits possible) ----
    v16bf pb[4];
    #pragma unroll
    for (int kt = 0; kt < 4; ++kt)
      pb[kt] = *(const v16bf*)(phB + (size_t)((mi * 2 + 1) * 4 + kt) * 512);

    v8f f0 = {}, f1 = {};
    #pragma unroll
    for (int kt = 0; kt < 4; ++kt) f0 = wmma_bf16(pa[kt], thB[kt], f0);
    #pragma unroll
    for (int kt = 0; kt < 4; ++kt) f1 = wmma_bf16(pb[kt], thB[kt], f1);

    // ---- exp (own batch only; max-free is safe: |f|/16 <~ 6) ----
    float e0[8], e1[8];
    #pragma unroll
    for (int r = 0; r < 8; ++r) e0[r] = __expf(f0[r] * 0.0625f);
    #pragma unroll
    for (int r = 0; r < 8; ++r) e1[r] = __expf(f1[r] * 0.0625f);

    // ---- exchange exp values: elem e = m*16+n, m = s*16 + r + 8*hiL, n = lm
    {
      int base0 = (8 * hiL * 16 + lm) * 10 + b;
      #pragma unroll
      for (int r = 0; r < 8; ++r) ebuf[base0 + r * 160] = e0[r];
      #pragma unroll
      for (int r = 0; r < 8; ++r) ebuf[base0 + 2560 + r * 160] = e1[r];
    }
    __syncthreads();

    // ---- cooperative reciprocal sums (2 elems/thread) ----
    {
      int e2 = t * 2;
      const float* p = ebuf + e2 * 10;
      float s0 = ((p[0] + p[1]) + (p[2] + p[3])) + ((p[4] + p[5]) + (p[6] + p[7]));
      float s1 = ((p[10] + p[11]) + (p[12] + p[13])) + ((p[14] + p[15]) + (p[16] + p[17]));
      int m = e2 >> 4, n = e2 & 15;
      sbuf[n * 34 + m]       = __frcp_rn(s0);
      sbuf[(n + 1) * 34 + m] = __frcp_rn(s1);
    }

    // prefetch next iteration's s=0 phi frags (overlaps phase3 + den WMMAs)
    if (mi < 127) {
      #pragma unroll
      for (int kt = 0; kt < 4; ++kt)
        pa[kt] = *(const v16bf*)(phB + (size_t)((mi * 2 + 2) * 4 + kt) * 512);
    }
    __syncthreads();

    // ---- build fnorm A-fragment fully in-lane (C-frag of f^T == A layout) ----
    union { v16bf v; unsigned u[8]; } fA;
    #pragma unroll
    for (int j = 0; j < 4; ++j) {
      float2 rs0 = *(const float2*)(sbuf + lm * 34 + (2 * j + 8 * hiL));
      float2 rs1 = *(const float2*)(sbuf + lm * 34 + (16 + 2 * j + 8 * hiL));
      fA.u[j]     = (unsigned)f2bf(e0[2 * j] * rs0.x) | ((unsigned)f2bf(e0[2 * j + 1] * rs0.y) << 16);
      fA.u[4 + j] = (unsigned)f2bf(e1[2 * j] * rs1.x) | ((unsigned)f2bf(e1[2 * j + 1] * rs1.y) << 16);
    }

    // ---- den accumulation, double-buffered x^T fragment loads ----
    v16bf bx = *(const v16bf*)(xtB + (size_t)(0 * 128 + mi) * 512);
    #pragma unroll
    for (int ct = 0; ct < 16; ++ct) {
      v16bf cur = bx;
      if (ct < 15) bx = *(const v16bf*)(xtB + (size_t)((ct + 1) * 128 + mi) * 512);
      dacc[ct] = wmma_bf16(fA.v, cur, dacc[ct]);
    }
    __syncthreads();
  }

  // ---- write den as [b][n][c] f32 (c-contiguous, coalesced 64B runs) ----
  #pragma unroll
  for (int ct = 0; ct < 16; ++ct)
    #pragma unroll
    for (int r = 0; r < 8; ++r) {
      int n = ntile * 16 + r + 8 * hiL;
      den_nc[(size_t)(b * 4096 + n) * 256 + ct * 16 + lm] = dacc[ct][r];
    }
}

// ============================================================================
// K3: out = den + w_conv . den + b_conv
// ============================================================================
__global__ __launch_bounds__(256) void k3_conv(
    const float* __restrict__ den_nc, const unsigned short* __restrict__ wcF,
    const float* __restrict__ b_conv, float* __restrict__ out)
{
  const int t = threadIdx.x, w = t >> 5, lane = t & 31;
  const int hi = lane >> 4, lm = lane & 15;
  const int b = blockIdx.x >> 8;
  const int n0 = (blockIdx.x & 255) * 16;

  v8f acc0 = {}, acc1 = {};
  const float* denrow = den_nc + (size_t)(b * 4096 + n0 + lm) * 256;

  #pragma unroll
  for (int ck = 0; ck < 8; ++ck) {
    union { v16bf v; unsigned short s[16]; } bf_;
    const float* src = denrow + ck * 32 + hi * 8;
    #pragma unroll
    for (int j = 0; j < 8; ++j) {
      float2 p = *(const float2*)(src + 2 * j + 8 * (j >> 2));
      bf_.s[2 * j]     = f2bf(p.x);
      bf_.s[2 * j + 1] = f2bf(p.y);
    }
    v16bf a0 = *(const v16bf*)(wcF + (size_t)((w * 8)       + ck) * 512 + lane * 16);
    v16bf a1 = *(const v16bf*)(wcF + (size_t)(((w + 8) * 8) + ck) * 512 + lane * 16);
    acc0 = wmma_bf16(a0, bf_.v, acc0);
    acc1 = wmma_bf16(a1, bf_.v, acc1);
  }

  const int n = n0 + lm;
  #pragma unroll
  for (int r = 0; r < 8; ++r) {
    int d0 = w * 16 + r + 8 * hi;
    int d1 = (w + 8) * 16 + r + 8 * hi;
    float r0 = den_nc[(size_t)(b * 4096 + n) * 256 + d0] + b_conv[d0] + acc0[r];
    float r1 = den_nc[(size_t)(b * 4096 + n) * 256 + d1] + b_conv[d1] + acc1[r];
    out[(size_t)(b * 256 + d0) * 4096 + n] = r0;
    out[(size_t)(b * 256 + d1) * 4096 + n] = r1;
  }
}

// ============================================================================
extern "C" void kernel_launch(void* const* d_in, const int* in_sizes, int n_in,
                              void* d_out, int out_size, void* d_ws, size_t ws_size,
                              hipStream_t stream) {
  const float* x       = (const float*)d_in[0];
  const float* w_theta = (const float*)d_in[1];
  const float* b_theta = (const float*)d_in[2];
  const float* w_phi   = (const float*)d_in[3];
  const float* b_phi   = (const float*)d_in[4];
  const float* w_conv  = (const float*)d_in[5];
  const float* b_conv  = (const float*)d_in[6];

  char* ws = (char*)d_ws;
  unsigned short* thF = (unsigned short*)(ws);                        //  8 MB (theta B-frags)
  unsigned short* phF = (unsigned short*)(ws + ((size_t)8  << 20));   //  8 MB (phi A-frags)
  unsigned short* xtF = (unsigned short*)(ws + ((size_t)16 << 20));   // 16 MB (x^T B-frags)
  unsigned short* wcF = (unsigned short*)(ws + ((size_t)32 << 20));   // 128 KB
  float*          den = (float*)        (ws + ((size_t)33 << 20));    // 32 MB

  k1_proj<<<1024, 256, 0, stream>>>(x, w_theta, b_theta, w_phi, b_phi, thF, phF);
  k1_xt  <<<2048, 256, 0, stream>>>(x, xtF);
  k1_wc  <<<32,   256, 0, stream>>>(w_conv, wcF);
  k2_attn<<<256,  256, 0, stream>>>(phF, thF, xtF, den);
  k3_conv<<<2048, 256, 0, stream>>>(den, wcF, b_conv, (float*)d_out);
}
